// deformable_fusion_42769284333652
// MI455X (gfx1250) — matrix operand order repaired and checked by reading, not compiled, not run on hardware
//
#include <hip/hip_runtime.h>
#include <hip/hip_bf16.h>

// ---------------------------------------------------------------------------
// Problem constants (fixed by the reference)
// ---------------------------------------------------------------------------
#define BS      2
#define EDIM    256
#define NQ      16384      // 128*128 queries
#define NV      21760      // 16384+4096+1024+256 value positions
#define NHEADS  8
#define HD      32         // head dim
#define NLVL    4
#define NPTS    4
#define HID     512
#define GROUPS  32
#define EPSV    1e-5f

static const int kLvlHW[NLVL]  = {16384, 4096, 1024, 256};
static const int kLvlOff[NLVL] = {0, 16384, 20480, 21504};
static const int kLvlC[NLVL]   = {24, 40, 112, 1280};

// ---------------------------------------------------------------------------
// WMMA fragment types (CDNA5 gfx1250, wave32)
// ---------------------------------------------------------------------------
typedef __attribute__((ext_vector_type(16))) __bf16 v16bf;
typedef __attribute__((ext_vector_type(8)))  float  v8f;

// pack two f32 -> two bf16 (truncate) with a single v_perm_b32
__device__ __forceinline__ unsigned pkbf(float lo, float hi) {
  return __builtin_amdgcn_perm(__float_as_uint(hi), __float_as_uint(lo), 0x07060302u);
}

union FragBF { v16bf v; unsigned u[8]; };

// A fragment (16x32 bf16): lane row = l16, K = half*16 + e.
// p = A + row*K + half*16 + k0  ->  4 aligned b128 loads at +0,+16,+32,+48 bytes
__device__ __forceinline__ v16bf load_a_frag(const float* __restrict__ p) {
  const float4 r0 = *(const float4*)(p);
  const float4 r1 = *(const float4*)(p + 4);
  const float4 r2 = *(const float4*)(p + 8);
  const float4 r3 = *(const float4*)(p + 12);
  FragBF f;
  f.u[0] = pkbf(r0.x, r0.y); f.u[1] = pkbf(r0.z, r0.w);
  f.u[2] = pkbf(r1.x, r1.y); f.u[3] = pkbf(r1.z, r1.w);
  f.u[4] = pkbf(r2.x, r2.y); f.u[5] = pkbf(r2.z, r2.w);
  f.u[6] = pkbf(r3.x, r3.y); f.u[7] = pkbf(r3.z, r3.w);
  return f.v;
}

// B fragment (32x16 bf16): lane col = l16, K = half*16 + e.
// p = B + (half*16)*LDB + col + k0*LDB -> 16 b32 loads at immediate offsets e*LDB*4
template<int LDB>
__device__ __forceinline__ v16bf load_b_frag(const float* __restrict__ p) {
  FragBF f;
#pragma unroll
  for (int j = 0; j < 8; ++j)
    f.u[j] = pkbf(p[(2 * j) * LDB], p[(2 * j + 1) * LDB]);
  return f.v;
}

// ---------------------------------------------------------------------------
// Guard-free GEMM:  C[b] = act( A[b] (MxK, row-major) * B (KxN=LDB) + bias )
// Requirements (guaranteed by the launcher): M%64==0, K%32==0, N==LDB.
// Block = 128 threads = 4 waves in 2x2; each wave owns a 32x64 tile
// (2 A-frags x 4 B-frags = 8 WMMA per k-step, A reused 4x).
// ---------------------------------------------------------------------------
template<int LDB>
__global__ __launch_bounds__(128)
void gemm_fast(const float* __restrict__ A, const float* __restrict__ B,
               const float* __restrict__ bias, float* __restrict__ C,
               int M, int K, long a_batch, long c_batch, int act) {
  const int lane = threadIdx.x & 31;
  const int wave = threadIdx.x >> 5;
  const int half = lane >> 4;
  const int l16  = lane & 15;
  const int bm   = blockIdx.y * 64 + (wave >> 1) * 32;
  const int bn   = blockIdx.x * 128 + (wave & 1) * 64;

  const float* Ab = A + (long)blockIdx.z * a_batch;
  float*       Cb = C + (long)blockIdx.z * c_batch;

  const float* pA0 = Ab + (long)(bm + l16) * K + half * 16;
  const float* pA1 = Ab + (long)(bm + 16 + l16) * K + half * 16;
  const float* pB0 = B + (long)(half * 16) * LDB + (bn + 0 * 16 + l16);
  const float* pB1 = B + (long)(half * 16) * LDB + (bn + 1 * 16 + l16);
  const float* pB2 = B + (long)(half * 16) * LDB + (bn + 2 * 16 + l16);
  const float* pB3 = B + (long)(half * 16) * LDB + (bn + 3 * 16 + l16);

  v8f acc[2][4] = {};

  for (int k0 = 0; k0 < K; k0 += 32) {
    const v16bf a0 = load_a_frag(pA0);
    const v16bf a1 = load_a_frag(pA1);
    const v16bf b0 = load_b_frag<LDB>(pB0);
    const v16bf b1 = load_b_frag<LDB>(pB1);
    const v16bf b2 = load_b_frag<LDB>(pB2);
    const v16bf b3 = load_b_frag<LDB>(pB3);
    pA0 += 32; pA1 += 32;
    pB0 += 32 * LDB; pB1 += 32 * LDB; pB2 += 32 * LDB; pB3 += 32 * LDB;

    acc[0][0] = __builtin_amdgcn_wmma_f32_16x16x32_bf16(false, a0, false, b0, (short)0, acc[0][0], false, false);
    acc[0][1] = __builtin_amdgcn_wmma_f32_16x16x32_bf16(false, a0, false, b1, (short)0, acc[0][1], false, false);
    acc[0][2] = __builtin_amdgcn_wmma_f32_16x16x32_bf16(false, a0, false, b2, (short)0, acc[0][2], false, false);
    acc[0][3] = __builtin_amdgcn_wmma_f32_16x16x32_bf16(false, a0, false, b3, (short)0, acc[0][3], false, false);
    acc[1][0] = __builtin_amdgcn_wmma_f32_16x16x32_bf16(false, a1, false, b0, (short)0, acc[1][0], false, false);
    acc[1][1] = __builtin_amdgcn_wmma_f32_16x16x32_bf16(false, a1, false, b1, (short)0, acc[1][1], false, false);
    acc[1][2] = __builtin_amdgcn_wmma_f32_16x16x32_bf16(false, a1, false, b2, (short)0, acc[1][2], false, false);
    acc[1][3] = __builtin_amdgcn_wmma_f32_16x16x32_bf16(false, a1, false, b3, (short)0, acc[1][3], false, false);
  }

  // C/D layout: VGPR r holds row (r + 8*half), col = lane&15.  No guards needed.
#pragma unroll
  for (int j = 0; j < 4; ++j) {
    const int nn = bn + j * 16 + l16;
    const float bvx = bias ? bias[nn] : 0.f;
#pragma unroll
    for (int i = 0; i < 2; ++i) {
#pragma unroll
      for (int r = 0; r < 8; ++r) {
        const int mm = bm + i * 16 + half * 8 + r;
        float v = acc[i][j][r] + bvx;
        if (act) v = fmaxf(v, 0.f);
        Cb[(long)mm * LDB + nn] = v;
      }
    }
  }
}

// ---------------------------------------------------------------------------
// NCHW conv input -> (b, HW, Cpad) zero-padded row-major (for guard-free GEMM)
// ---------------------------------------------------------------------------
__global__ __launch_bounds__(256)
void feat_pad_transpose(const float* __restrict__ x, float* __restrict__ out,
                        int C, int HW, int Cpad) {
  const long i = (long)blockIdx.x * blockDim.x + threadIdx.x;
  const long total = (long)BS * HW * Cpad;
  if (i >= total) return;
  const int  c  = (int)(i % Cpad);
  const long t  = i / Cpad;
  const int  hw = (int)(t % HW);
  const int  b  = (int)(t / HW);
  out[i] = (c < C) ? x[((long)b * C + c) * HW + hw] : 0.f;
}

// proj_W (256, C) -> (Cpad, 256) zero-padded row-major
__global__ __launch_bounds__(256)
void weight_pad_transpose(const float* __restrict__ w, float* __restrict__ out,
                          int C, int Cpad) {
  const int i = blockIdx.x * blockDim.x + threadIdx.x;
  if (i >= Cpad * 256) return;
  const int o = i & 255;
  const int k = i >> 8;
  out[i] = (k < C) ? w[o * C + k] : 0.f;
}

// ---------------------------------------------------------------------------
// GroupNorm over one (batch, group): reduce HW x 8 channels, normalize inplace.
// flat layout: (b, NV, 256) at level offset.
// ---------------------------------------------------------------------------
__global__ __launch_bounds__(256)
void groupnorm_kernel(float* __restrict__ flat,
                      const float* __restrict__ gamma,
                      const float* __restrict__ beta,
                      int HW, int posOff) {
  const int g = blockIdx.x;
  const int b = blockIdx.y;
  float* base = flat + ((long)b * NV + posOff) * EDIM + g * 8;
  const int n = HW * 8;

  float s = 0.f, ss = 0.f;
  for (int i = threadIdx.x; i < n; i += blockDim.x) {
    const int hw = i >> 3, c = i & 7;
    const float v = base[(long)hw * EDIM + c];
    s += v; ss += v * v;
  }
  __shared__ float sh0[256], sh1[256];
  sh0[threadIdx.x] = s; sh1[threadIdx.x] = ss;
  __syncthreads();
  for (int st = 128; st > 0; st >>= 1) {
    if ((int)threadIdx.x < st) {
      sh0[threadIdx.x] += sh0[threadIdx.x + st];
      sh1[threadIdx.x] += sh1[threadIdx.x + st];
    }
    __syncthreads();
  }
  const float inv_n = 1.f / (float)n;
  const float mu  = sh0[0] * inv_n;
  const float var = sh1[0] * inv_n - mu * mu;
  const float rs  = rsqrtf(var + EPSV);

  for (int i = threadIdx.x; i < n; i += blockDim.x) {
    const int hw = i >> 3, c = i & 7;
    const long idx = (long)hw * EDIM + c;
    base[idx] = (base[idx] - mu) * rs * gamma[g * 8 + c] + beta[g * 8 + c];
  }
}

// ---------------------------------------------------------------------------
// Softmax over the 16 (level,point) logits per (b,q,head), in place.
// ---------------------------------------------------------------------------
__global__ __launch_bounds__(256)
void softmax16_kernel(float* __restrict__ aw, long n) {
  const long t = (long)blockIdx.x * blockDim.x + threadIdx.x;
  if (t >= n) return;
  const long row  = t >> 3;
  const int  head = (int)(t & 7);
  float* p = aw + row * (NHEADS * NLVL * NPTS) + head * (NLVL * NPTS);
  float v[16], mx = -3.4e38f;
#pragma unroll
  for (int i = 0; i < 16; ++i) { v[i] = p[i]; mx = fmaxf(mx, v[i]); }
  float s = 0.f;
#pragma unroll
  for (int i = 0; i < 16; ++i) { v[i] = expf(v[i] - mx); s += v[i]; }
  const float inv = 1.f / s;
#pragma unroll
  for (int i = 0; i < 16; ++i) p[i] = v[i] * inv;
}

// ---------------------------------------------------------------------------
// Deformable sampling: one wave32 per (b, q, head); lane = channel within head.
// Branch-free corners: always load clamped address (coalesced 128B per corner),
// zero the *weight* when out of bounds — EXEC stays uniform.
// ---------------------------------------------------------------------------
__global__ __launch_bounds__(256)
void msda_sample_kernel(const float* __restrict__ v,
                        const float* __restrict__ offs,
                        const float* __restrict__ aw,
                        float* __restrict__ out) {
  const int gw   = (int)((blockIdx.x * blockDim.x + threadIdx.x) >> 5);
  const int lane = threadIdx.x & 31;
  const int head = gw & 7;
  const int q    = (gw >> 3) & (NQ - 1);
  const int b    = gw >> 17;

  const float refx = (float)(q & 127) * (1.f / 127.f);
  const float refy = (float)(q >> 7)  * (1.f / 127.f);

  const float* offq = offs + ((long)(b * NQ + q)) * EDIM + head * (NLVL * NPTS * 2);
  const float* awq  = aw   + ((long)(b * NQ + q)) * (NHEADS * NLVL * NPTS) + head * (NLVL * NPTS);
  const float* vb   = v    + (long)b * NV * EDIM + head * HD + lane;

  float acc = 0.f;
#pragma unroll
  for (int lvl = 0; lvl < NLVL; ++lvl) {
    const int W = (lvl == 0) ? 128 : (lvl == 1) ? 64 : (lvl == 2) ? 32 : 16;
    const int H = W;
    const int base = (lvl == 0) ? 0 : (lvl == 1) ? 16384 : (lvl == 2) ? 20480 : 21504;
#pragma unroll
    for (int p = 0; p < NPTS; ++p) {
      const float ox = offq[(lvl * NPTS + p) * 2 + 0];
      const float oy = offq[(lvl * NPTS + p) * 2 + 1];
      const float a  = awq[lvl * NPTS + p];
      const float x = refx * (float)W + ox - 0.5f;
      const float y = refy * (float)H + oy - 0.5f;
      const float x0f = floorf(x), y0f = floorf(y);
      const int x0 = (int)x0f, y0 = (int)y0f;
      const float wx1 = x - x0f, wy1 = y - y0f;
      const float wx0 = 1.f - wx1, wy0 = 1.f - wy1;

      const int xs[4] = {x0, x0 + 1, x0, x0 + 1};
      const int ys[4] = {y0, y0, y0 + 1, y0 + 1};
      const float wgt[4] = {wx0 * wy0, wx1 * wy0, wx0 * wy1, wx1 * wy1};
#pragma unroll
      for (int c = 0; c < 4; ++c) {
        const int xi = xs[c], yi = ys[c];
        const int xc = (xi < 0) ? 0 : (xi > W - 1 ? W - 1 : xi);
        const int yc = (yi < 0) ? 0 : (yi > H - 1 ? H - 1 : yi);
        const bool valid = (xi >= 0) & (xi < W) & (yi >= 0) & (yi < H);
        const float val = vb[((long)(base + yc * W + xc)) * EDIM];  // always issued
        acc += a * (valid ? wgt[c] : 0.f) * val;
      }
    }
  }
  out[((long)(b * NQ + q)) * EDIM + head * HD + lane] = acc;
}

// ---------------------------------------------------------------------------
// fused = sigmoid(alpha)*sat + (1-sigmoid(alpha))*osm + q
// ---------------------------------------------------------------------------
__global__ __launch_bounds__(256)
void fuse_kernel(const float* __restrict__ q, const float* __restrict__ so,
                 const float* __restrict__ oo, const float* __restrict__ alpha_p,
                 float* __restrict__ out, long n) {
  const long i = (long)blockIdx.x * blockDim.x + threadIdx.x;
  if (i >= n) return;
  const float a = 1.f / (1.f + expf(-alpha_p[0]));
  out[i] = a * so[i] + (1.f - a) * oo[i] + q[i];
}

// ---------------------------------------------------------------------------
// out = fused + LayerNorm(h2) * g + b   — one wave32 per row of 256.
// ---------------------------------------------------------------------------
__global__ __launch_bounds__(256)
void ln_residual_kernel(const float* __restrict__ fused,
                        const float* __restrict__ h2,
                        const float* __restrict__ g,
                        const float* __restrict__ bta,
                        float* __restrict__ out, int rows) {
  const int wid  = (int)((blockIdx.x * blockDim.x + threadIdx.x) >> 5);
  const int lane = threadIdx.x & 31;
  if (wid >= rows) return;
  const float* hr = h2 + (long)wid * EDIM;
  float v[8];
  float s = 0.f;
#pragma unroll
  for (int i = 0; i < 8; ++i) { v[i] = hr[lane * 8 + i]; s += v[i]; }
#pragma unroll
  for (int m = 16; m > 0; m >>= 1) s += __shfl_xor(s, m, 32);
  const float mu = s * (1.f / 256.f);
  float ss = 0.f;
#pragma unroll
  for (int i = 0; i < 8; ++i) { const float d = v[i] - mu; ss += d * d; }
#pragma unroll
  for (int m = 16; m > 0; m >>= 1) ss += __shfl_xor(ss, m, 32);
  const float rs = rsqrtf(ss * (1.f / 256.f) + EPSV);
#pragma unroll
  for (int i = 0; i < 8; ++i) {
    const int ch = lane * 8 + i;
    out[(long)wid * EDIM + ch] =
        fused[(long)wid * EDIM + ch] + (v[i] - mu) * rs * g[ch] + bta[ch];
  }
}

// ---------------------------------------------------------------------------
// Host side
// ---------------------------------------------------------------------------
struct ModParams {
  const float *proj_W[4], *proj_b[4], *gn_g[4], *gn_b[4];
  const float *Wv, *bv, *Wo, *bo, *Wa, *ba, *Wout, *bout;
};
struct FfnParams { const float *W1, *b1, *W2, *b2, *ln_g, *ln_b; };

static int parse_mod(void* const* d_in, int idx, bool sorted, ModParams* p) {
  auto F = [&](int i) { return (const float*)d_in[i]; };
  if (sorted) {  // jax pytree order: gn_b, gn_g, msda{Wa,Wo,Wout,Wv,ba,bo,bout,bv}, proj_W, proj_b
    for (int i = 0; i < 4; ++i) p->gn_b[i] = F(idx++);
    for (int i = 0; i < 4; ++i) p->gn_g[i] = F(idx++);
    p->Wa = F(idx++); p->Wo = F(idx++); p->Wout = F(idx++); p->Wv = F(idx++);
    p->ba = F(idx++); p->bo = F(idx++); p->bout = F(idx++); p->bv = F(idx++);
    for (int i = 0; i < 4; ++i) p->proj_W[i] = F(idx++);
    for (int i = 0; i < 4; ++i) p->proj_b[i] = F(idx++);
  } else {       // dict insertion order
    for (int i = 0; i < 4; ++i) p->proj_W[i] = F(idx++);
    for (int i = 0; i < 4; ++i) p->proj_b[i] = F(idx++);
    for (int i = 0; i < 4; ++i) p->gn_g[i] = F(idx++);
    for (int i = 0; i < 4; ++i) p->gn_b[i] = F(idx++);
    p->Wv = F(idx++); p->bv = F(idx++); p->Wo = F(idx++); p->bo = F(idx++);
    p->Wa = F(idx++); p->ba = F(idx++); p->Wout = F(idx++); p->bout = F(idx++);
  }
  return idx;
}

// N must be one of 128/256/512; M%64==0, K%32==0.
static void launch_gemm(hipStream_t s, int N,
                        const float* A, const float* B, const float* bias,
                        float* C, int M, int K, long a_batch, long c_batch, int act) {
  dim3 grid(N / 128 > 0 ? N / 128 : 1, M / 64, BS);
  if (N == 128)
    gemm_fast<128><<<dim3(1, M / 64, BS), dim3(128), 0, s>>>(A, B, bias, C, M, K, a_batch, c_batch, act);
  else if (N == 256)
    gemm_fast<256><<<dim3(2, M / 64, BS), dim3(128), 0, s>>>(A, B, bias, C, M, K, a_batch, c_batch, act);
  else
    gemm_fast<512><<<dim3(4, M / 64, BS), dim3(128), 0, s>>>(A, B, bias, C, M, K, a_batch, c_batch, act);
}

extern "C" void kernel_launch(void* const* d_in, const int* in_sizes, int n_in,
                              void* d_out, int out_size, void* d_ws, size_t ws_size,
                              hipStream_t stream) {
  const float* q = (const float*)d_in[0];
  const float* feats[2][4];
  for (int i = 0; i < 4; ++i) feats[0][i] = (const float*)d_in[1 + i]; // sat
  for (int i = 0; i < 4; ++i) feats[1][i] = (const float*)d_in[5 + i]; // osm

  // Detect flattening convention of the param dicts via in_sizes:
  // pytree-sorted => first leaf is gn_b[0] (256); insertion => proj_W[0] (6144)
  const bool sorted = (in_sizes[9] == 256);
  ModParams mp[2];
  int idx = 9;
  idx = parse_mod(d_in, idx, sorted, &mp[0]);
  idx = parse_mod(d_in, idx, sorted, &mp[1]);
  FfnParams ffn;
  {
    auto F = [&](int i) { return (const float*)d_in[i]; };
    if (sorted) { ffn.W1 = F(idx++); ffn.W2 = F(idx++); ffn.b1 = F(idx++);
                  ffn.b2 = F(idx++); ffn.ln_b = F(idx++); ffn.ln_g = F(idx++); }
    else        { ffn.W1 = F(idx++); ffn.b1 = F(idx++); ffn.W2 = F(idx++);
                  ffn.b2 = F(idx++); ffn.ln_g = F(idx++); ffn.ln_b = F(idx++); }
  }
  const float* alpha_p = (const float*)d_in[idx];

  // ---- workspace layout (floats), with phase reuse ----
  float* ws = (float*)d_ws;
  const long N_FLAT = (long)BS * NV * EDIM;      // 11,141,120
  const long N_ROW  = (long)BS * NQ * EDIM;      //  8,388,608
  const long N_AW   = (long)BS * NQ * 128;       //  4,194,304
  float* FLAT  = ws;                             // conv+GN output (also FUSED later)
  float* V     = FLAT + N_FLAT;                  // projected values (also H later)
  float* OFFS  = V + N_FLAT;                     // sampling offsets
  float* AW    = OFFS + N_ROW;                   // attention weights
  float* MSDA  = AW + N_AW;                      // sampled heads (also FFN2 out later)
  float* SOUT  = MSDA + N_ROW;                   // sat branch output
  float* OOUT  = SOUT + N_ROW;                   // osm branch output
  float* XFEAT = OOUT + N_ROW;                   // padded/transposed conv input (<= 1,048,576)
  float* WT    = XFEAT + 1048576;                // padded/transposed conv weight (<= 327,680)
  float* FUSED = FLAT;                           // reuse
  float* H     = V;                              // reuse (16.8M fits in V+OFFS span)
  float* H2    = MSDA;                           // reuse

  float* branch_out[2] = {SOUT, OOUT};

  for (int mod = 0; mod < 2; ++mod) {
    const ModParams& P = mp[mod];

    // 1x1 conv per level via pad/transpose + guard-free WMMA GEMM
    for (int l = 0; l < NLVL; ++l) {
      const int HW = kLvlHW[l];
      const int C  = kLvlC[l];
      const int Cpad = (C + 31) & ~31;
      weight_pad_transpose<<<dim3((Cpad * 256 + 255) / 256), dim3(256), 0, stream>>>(
          P.proj_W[l], WT, C, Cpad);
      const long totT = (long)BS * HW * Cpad;
      feat_pad_transpose<<<dim3((int)((totT + 255) / 256)), dim3(256), 0, stream>>>(
          feats[mod][l], XFEAT, C, HW, Cpad);
      launch_gemm(stream, EDIM, XFEAT, WT, P.proj_b[l],
                  FLAT + (long)kLvlOff[l] * EDIM, HW, Cpad,
                  (long)HW * Cpad, (long)NV * EDIM, 0);
    }
    // GroupNorm in place
    for (int l = 0; l < NLVL; ++l) {
      groupnorm_kernel<<<dim3(GROUPS, BS), dim3(256), 0, stream>>>(
          FLAT, P.gn_g[l], P.gn_b[l], kLvlHW[l], kLvlOff[l]);
    }
    // value projection: V = FLAT @ Wv + bv
    launch_gemm(stream, EDIM, FLAT, P.Wv, P.bv, V, NV, EDIM,
                (long)NV * EDIM, (long)NV * EDIM, 0);
    // offsets: OFFS = q @ Wo + bo
    launch_gemm(stream, EDIM, q, P.Wo, P.bo, OFFS, NQ, EDIM,
                (long)NQ * EDIM, (long)NQ * EDIM, 0);
    // attention logits: AW = q @ Wa + ba, then softmax over 16
    launch_gemm(stream, 128, q, P.Wa, P.ba, AW, NQ, EDIM,
                (long)NQ * EDIM, (long)NQ * 128, 0);
    softmax16_kernel<<<dim3((BS * NQ * NHEADS + 255) / 256), dim3(256), 0, stream>>>(
        AW, (long)BS * NQ * NHEADS);
    // deformable sampling -> MSDA (b,q,256)
    msda_sample_kernel<<<dim3(BS * NQ * NHEADS / 8), dim3(256), 0, stream>>>(
        V, OFFS, AW, MSDA);
    // output projection
    launch_gemm(stream, EDIM, MSDA, P.Wout, P.bout, branch_out[mod], NQ, EDIM,
                (long)NQ * EDIM, (long)NQ * EDIM, 0);
  }

  // fused = a*sat + (1-a)*osm + q
  fuse_kernel<<<dim3((int)(N_ROW / 256)), dim3(256), 0, stream>>>(
      q, SOUT, OOUT, alpha_p, FUSED, N_ROW);

  // FFN: H = relu(FUSED @ W1 + b1); H2 = H @ W2 + b2
  launch_gemm(stream, HID, FUSED, ffn.W1, ffn.b1, H, NQ, EDIM,
              (long)NQ * EDIM, (long)NQ * HID, 1);
  launch_gemm(stream, EDIM, H, ffn.W2, ffn.b2, H2, NQ, HID,
              (long)NQ * HID, (long)NQ * EDIM, 0);

  // out = FUSED + LN(H2)
  ln_residual_kernel<<<dim3(BS * NQ / 8), dim3(256), 0, stream>>>(
      FUSED, H2, ffn.ln_g, ffn.ln_b, (float*)d_out, BS * NQ);
}
